// MamHDR_31190052503722
// MI455X (gfx1250) — matrix operand order repaired and checked by reading, not compiled, not run on hardware
//
#include <hip/hip_runtime.h>

// ---------------------------------------------------------------------------
// MamHDR forward for MI455X (gfx1250, wave32, WMMA).
// All channel-mixing matmuls go through v_wmma_f32_16x16x32_f16 (f16 in,
// f32 accumulate). Activations kept NCHW fp32. Selective scan runs one
// thread per (direction, channel) with 16 SSM states in registers and
// B/C vectors double-buffered in LDS (one barrier per step).
// ---------------------------------------------------------------------------

#define B_   2
#define DM_  64
#define DI_  128
#define NS_  16
#define RK_  4
#define H_   64
#define W_   64
#define L_   4096   // H_*W_

typedef __attribute__((ext_vector_type(16))) _Float16 v16h;
typedef __attribute__((ext_vector_type(8)))  float    v8f;

__device__ __forceinline__ float mh_sigm(float v) {
  // v_rcp_f32 instead of IEEE divide: activation precision is ample.
  return __builtin_amdgcn_rcpf(1.f + __expf(-v));
}
__device__ __forceinline__ float mh_act(float v, int act, float slope) {
  if (act == 1) return v >= 0.f ? v : slope * v;            // leaky relu
  if (act == 2) return v * mh_sigm(v);                      // silu
  if (act == 3) return mh_sigm(v);                          // sigmoid
  return v;
}

// ---------------------------------------------------------------------------
// WMMA GEMM: Y[z][m][n] = act( sum_k W[..] * X[z][k][n] + bias[m] )
//   WT=0: W[k*wld + m]   (weight stored [Cin, Cout] row-major, e.g. x @ W)
//   WT=1: W[m*wld + k]   (weight stored [Cout, Cin], e.g. 1x1 conv OIHW)
// GUARD=0 fast path requires Cout%16==0 and Cin%32==0 (no bounds checks in
// the inner loop; OOB M-tiles exit with a wave-uniform early return).
// GUARD=1 path uses clamped addresses + value select (branch-free loads).
// Row length of X/Y is always L_ (pixels). blockIdx.z batches X/Y by
// xbs/ybs and (optionally, nw>1) the weights by wbs.
// Each wave computes one 16x16 (MxN) tile, K in steps of 32.
// ---------------------------------------------------------------------------
template<int WT, int GUARD>
__global__ __launch_bounds__(256) void mh_gemm_wmma(
    const float* __restrict__ X, const float* __restrict__ W,
    const float* __restrict__ bias, float* __restrict__ Y,
    int Cin, int Cout, int wld, long xbs, long ybs, long wbs, int nw,
    int act, float slope)
{
  const int  lane = threadIdx.x;      // 0..31
  const int  lh   = lane & 15;
  const bool hiH  = lane >= 16;
  const int  m0   = blockIdx.x * 64 + threadIdx.y * 16;
  const int  n0   = blockIdx.y * 32 + threadIdx.z * 16;
  if (!GUARD && m0 >= Cout) return;   // wave-uniform; no LDS/barriers in kernel
  const float* Xz = X + (long)blockIdx.z * xbs;
  float*       Yz = Y + (long)blockIdx.z * ybs;
  const float* Wz = W + (long)(blockIdx.z % nw) * wbs;

  v8f acc = {};
  for (int k0 = 0; k0 < Cin; k0 += 32) {
    v16h a, b;
    if (GUARD) {
#pragma unroll
      for (int i = 0; i < 16; ++i) {
        const int p  = i >> 1, o = i & 1;
        const int ka = (hiH ? 16 : 0) + 2 * p + o;
        const int kk = k0 + ka;
        const int m  = m0 + lh;
        // clamp addresses (always in-bounds) and select value to 0: the
        // loads stay unconditional -> no exec-mask churn, one wait.
        const int kks = kk < Cin ? kk : 0;
        const int ms  = m < Cout ? m : 0;
        float wv = WT ? Wz[(size_t)ms * wld + kks] : Wz[(size_t)kks * wld + ms];
        if (m >= Cout || kk >= Cin) wv = 0.f;
        a[i] = (_Float16)wv;
        const int kb  = k0 + (hiH ? 16 : 0) + i;
        const int kbs = kb < Cin ? kb : 0;
        float xv = Xz[(size_t)kbs * L_ + n0 + lh];
        if (kb >= Cin) xv = 0.f;
        b[i] = (_Float16)xv;
      }
    } else {
      // Fast path: fully unguarded fragment loads (clause-friendly).
      const float* Wp = WT ? (Wz + (size_t)(m0 + lh) * wld + k0)
                           : (Wz + (size_t)k0 * wld + (m0 + lh));
      const float* Xp = Xz + (size_t)(k0 + (hiH ? 16 : 0)) * L_ + n0 + lh;
      if (k0 + 32 < Cin)
        __builtin_prefetch(Xp + (size_t)32 * L_, 0, 0);   // global_prefetch_b8
#pragma unroll
      for (int i = 0; i < 16; ++i) {
        const int p  = i >> 1, o = i & 1;
        const int ka = (hiH ? 16 : 0) + 2 * p + o;
        a[i] = (_Float16)(WT ? Wp[ka] : Wp[(size_t)ka * wld]);
        b[i] = (_Float16)Xp[(size_t)i * L_];
      }
    }
    acc = __builtin_amdgcn_wmma_f32_16x16x32_f16(false, a, false, b,
                                                 (short)0, acc, false, false);
  }
  // D layout: VGPR r -> M = r (+8 for lanes 16-31), N = lane%16
#pragma unroll
  for (int r = 0; r < 8; ++r) {
    const int m = m0 + (hiH ? 8 : 0) + r;
    if (GUARD && m >= Cout) continue;
    float v = acc[r];
    if (bias) v += bias[m];
    Yz[(size_t)m * L_ + n0 + lh] = mh_act(v, act, slope);
  }
}

// ---------------------------------------------------------------------------
// Generic direct conv (3x3, grouped / dilated / depthwise). OIHW weights.
// ---------------------------------------------------------------------------
__global__ void mh_conv2d(const float* __restrict__ x, const float* __restrict__ w,
                          const float* __restrict__ bias, float* __restrict__ y,
                          int Cin, int Cout, int K, int pad, int dil, int groups,
                          int act, float slope)
{
  long idx = (long)blockIdx.x * blockDim.x + threadIdx.x;
  long total = (long)B_ * Cout * L_;
  if (idx >= total) return;
  int l = (int)(idx % L_); long t = idx / L_;
  int co = (int)(t % Cout); int bb = (int)(t / Cout);
  int wq = l % W_, hq = l / W_;
  int cpg = Cin / groups, opg = Cout / groups;
  int g = co / opg;
  float s = bias ? bias[co] : 0.f;
  const float* xb = x + ((long)bb * Cin + (long)g * cpg) * L_;
  const float* wb = w + (long)co * cpg * K * K;
  for (int ci = 0; ci < cpg; ++ci)
    for (int kh = 0; kh < K; ++kh) {
      int ih = hq + kh * dil - pad;
      if (ih < 0 || ih >= H_) continue;
      for (int kw = 0; kw < K; ++kw) {
        int iw = wq + kw * dil - pad;
        if (iw < 0 || iw >= W_) continue;
        s += xb[(long)ci * L_ + ih * W_ + iw] * wb[(ci * K + kh) * K + kw];
      }
    }
  y[idx] = mh_act(s, act, slope);
}

// LayerNorm over the channel axis of NCHW (== ln2d and ln_last in NCHW).
__global__ void mh_ln_ch(const float* __restrict__ x, const float* __restrict__ g,
                         const float* __restrict__ b, float* __restrict__ y,
                         int C, float eps)
{
  long p = (long)blockIdx.x * blockDim.x + threadIdx.x;
  if (p >= (long)B_ * L_) return;
  int bb = (int)(p / L_), l = (int)(p % L_);
  const float* xp = x + (long)bb * C * L_ + l;
  float* yp = y + (long)bb * C * L_ + l;
  float mu = 0.f;
  for (int c = 0; c < C; ++c) mu += xp[(long)c * L_];
  mu /= C;
  float v = 0.f;
  for (int c = 0; c < C; ++c) { float d = xp[(long)c * L_] - mu; v += d * d; }
  float r = rsqrtf(v / C + eps);
  for (int c = 0; c < C; ++c)
    yp[(long)c * L_] = (xp[(long)c * L_] - mu) * r * g[c] + b[c];
}

// ---- elementwise helpers ----
__global__ void mh_add(const float* a, const float* b, float* y, long n) {
  long i = (long)blockIdx.x * blockDim.x + threadIdx.x;
  if (i < n) y[i] = a[i] + b[i];
}
__global__ void mh_copy(const float* a, float* y, long n) {
  long i = (long)blockIdx.x * blockDim.x + threadIdx.x;
  if (i < n) y[i] = a[i];
}
__global__ void mh_addscale(const float* a, const float* b, const float* s,
                            float* y, int C, long n) {
  long i = (long)blockIdx.x * blockDim.x + threadIdx.x;
  if (i >= n) return;
  int c = (int)((i / L_) % C);
  y[i] = a[i] + b[i] * s[c];
}
__global__ void mh_mul_silu(const float* a, const float* z, float* y, long n) {
  long i = (long)blockIdx.x * blockDim.x + threadIdx.x;
  if (i >= n) return;
  float zz = z[i];
  y[i] = a[i] * zz * __builtin_amdgcn_rcpf(1.f + __expf(-zz));
}
__global__ void mh_halfmul(const float* x, float* y) {   // (B,128,L) -> (B,64,L): x1*x2
  long idx = (long)blockIdx.x * blockDim.x + threadIdx.x;
  if (idx >= (long)B_ * DM_ * L_) return;
  int l = (int)(idx % L_); long t = idx / L_;
  int c = (int)(t % DM_); int bb = (int)(t / DM_);
  const float* xb = x + (long)bb * 128 * L_;
  y[idx] = xb[(long)c * L_ + l] * xb[(long)(c + DM_) * L_ + l];
}
__global__ void mh_mul_bc(float* x, const float* s, int C) { // x[b,c,:] *= s[b,c]
  long idx = (long)blockIdx.x * blockDim.x + threadIdx.x;
  if (idx >= (long)B_ * C * L_) return;
  int c = (int)((idx / L_) % C); int bb = (int)(idx / ((long)C * L_));
  x[idx] *= s[bb * C + c];
}
__global__ void mh_concat2(const float* a, const float* b, float* y) { // 64+64 -> 128
  long idx = (long)blockIdx.x * blockDim.x + threadIdx.x;
  if (idx >= (long)B_ * 128 * L_) return;
  int l = (int)(idx % L_); long t = idx / L_;
  int c = (int)(t % 128); int bb = (int)(t / 128);
  y[idx] = (c < 64) ? a[((long)bb * 64 + c) * L_ + l]
                    : b[((long)bb * 64 + (c - 64)) * L_ + l];
}
__global__ void mh_cat3(const float* f1c, const float* a12, const float* f2c,
                        const float* f3c, const float* a32, float* y) {
  long idx = (long)blockIdx.x * blockDim.x + threadIdx.x;
  if (idx >= (long)B_ * 192 * L_) return;
  int l = (int)(idx % L_); long t = idx / L_;
  int c = (int)(t % 192); int bb = (int)(t / 192);
  long p = ((long)bb * 64 + (c % 64)) * L_ + l;
  float v;
  if (c < 64)       v = f1c[p] * a12[p];
  else if (c < 128) v = f2c[p];
  else              v = f3c[p] * a32[p];
  y[idx] = v;
}

// Build 4 scan directions from xc (B,DI,L): hw, wh, rev-hw, rev-wh.
__global__ void mh_build_xs(const float* __restrict__ xc, float* __restrict__ xs) {
  long idx = (long)blockIdx.x * blockDim.x + threadIdx.x;
  if (idx >= (long)B_ * 4 * DI_ * L_) return;
  int l = (int)(idx % L_); long t = idx / L_;
  int d = (int)(t % DI_); t /= DI_;
  int k = (int)(t % 4);  int bb = (int)(t / 4);
  int src;
  if (k == 0)      src = l;
  else if (k == 2) src = L_ - 1 - l;
  else {
    int ll = (k == 1) ? l : (L_ - 1 - l);
    src = (ll % H_) * W_ + (ll / H_);     // wh scan index ll = w*H + h
  }
  xs[idx] = xc[((long)bb * DI_ + d) * L_ + src];
}

// Selective scan: one block per (b, direction), one thread per channel d.
// dbl rows: 0..3 = dtr, 4..19 = B, 20..35 = C. ys written in place over xs.
__global__ __launch_bounds__(128) void mh_scan(
    float* xs, const float* __restrict__ dbl,
    const float* __restrict__ dt_w, const float* __restrict__ dt_b,
    const float* __restrict__ A_log, const float* __restrict__ Dp)
{
  const int z = blockIdx.x;           // b*4 + k
  const int k = z & 3;
  const int d = threadIdx.x;          // 0..127
  float* xz = xs + (long)z * DI_ * L_;
  const float* dz = dbl + (long)z * 36 * L_;

  float A[NS_], h[NS_];
#pragma unroll
  for (int n = 0; n < NS_; ++n) {
    A[n] = -__expf(A_log[((long)k * DI_ + d) * NS_ + n]);
    h[n] = 0.f;
  }
  const float w0 = dt_w[((long)k * DI_ + d) * RK_ + 0];
  const float w1 = dt_w[((long)k * DI_ + d) * RK_ + 1];
  const float w2 = dt_w[((long)k * DI_ + d) * RK_ + 2];
  const float w3 = dt_w[((long)k * DI_ + d) * RK_ + 3];
  const float db = dt_b[(long)k * DI_ + d];
  const float Dv = Dp[(long)k * DI_ + d];

  __shared__ float Bs[2][NS_], Cs[2][NS_];
  for (int l = 0; l < L_; ++l) {
    const int buf = l & 1;
    if (d < NS_)            Bs[buf][d]       = dz[(long)(4 + d) * L_ + l];
    else if (d < 2 * NS_)   Cs[buf][d - NS_] = dz[(long)(20 + d - NS_) * L_ + l];
    __syncthreads();   // single barrier per step; double buffer covers WAR
    float dtv = w0 * dz[l] + w1 * dz[(long)L_ + l] +
                w2 * dz[(long)2 * L_ + l] + w3 * dz[(long)3 * L_ + l] + db;
    dtv = (dtv > 20.f) ? dtv : log1pf(__expf(dtv));   // softplus
    const float u  = xz[(long)d * L_ + l];
    const float du = dtv * u;
    float acc = 0.f;
#pragma unroll
    for (int n = 0; n < NS_; ++n) {
      h[n] = h[n] * __expf(dtv * A[n]) + du * Bs[buf][n];
      acc += h[n] * Cs[buf][n];
    }
    xz[(long)d * L_ + l] = acc + u * Dv;
  }
}

// Merge the 4 directional scan outputs back to (B,DI,L) in hw order.
__global__ void mh_combine(const float* __restrict__ ys, float* __restrict__ y) {
  long idx = (long)blockIdx.x * blockDim.x + threadIdx.x;
  if (idx >= (long)B_ * DI_ * L_) return;
  int l = (int)(idx % L_); long t = idx / L_;
  int d = (int)(t % DI_); int bb = (int)(t / DI_);
  int hq = l / W_, wq = l % W_;
  int lwh = wq * H_ + hq;
  const float* base = ys + (long)bb * 4 * DI_ * L_;
  const long cs = (long)DI_ * L_;
  y[idx] = base[(long)d * L_ + l]
         + base[2 * cs + (long)d * L_ + (L_ - 1 - l)]
         + base[cs + (long)d * L_ + lwh]
         + base[3 * cs + (long)d * L_ + (L_ - 1 - lwh)];
}

// Spatial mean per (b,c) and tiny 64x64 linear for the SCA branch.
__global__ void mh_smean(const float* __restrict__ x, float* __restrict__ out) {
  __shared__ float sm[256];
  long base = (long)blockIdx.x * L_;
  float s = 0.f;
  for (int i = threadIdx.x; i < L_; i += 256) s += x[base + i];
  sm[threadIdx.x] = s; __syncthreads();
  for (int st = 128; st > 0; st >>= 1) {
    if ((int)threadIdx.x < st) sm[threadIdx.x] += sm[threadIdx.x + st];
    __syncthreads();
  }
  if (threadIdx.x == 0) out[blockIdx.x] = sm[0] * (1.f / (float)L_);
}
__global__ void mh_slinear(const float* __restrict__ in, const float* __restrict__ w,
                           const float* __restrict__ b, float* __restrict__ out) {
  int t = threadIdx.x;
  if (t >= B_ * DM_) return;
  int bb = t / DM_, co = t % DM_;
  float s = b[co];
  for (int ci = 0; ci < DM_; ++ci) s += w[co * DM_ + ci] * in[bb * DM_ + ci];
  out[t] = s;
}

// ---------------------------------------------------------------------------
// Host-side wiring
// ---------------------------------------------------------------------------
struct SSP {
  const float *in_w, *conv_w, *conv_b, *xproj_w, *dt_w, *dt_b, *A_log, *D,
              *on_g, *on_b, *out_w;
};
struct BLK {
  const float *ln_in_g, *ln_in_b;
  SSP ss;
  const float *ln_ffn_g, *ln_ffn_b, *fc1_w, *fc1_b, *fc2_w, *fc2_b,
              *scale1, *scale2;
};
struct DCP { const float *w1,*b1,*w2,*b2,*w3,*b3,*w4,*b4,*w5,*b5; };

extern "C" void kernel_launch(void* const* d_in, const int* in_sizes, int n_in,
                              void* d_out, int out_size, void* d_ws, size_t ws_size,
                              hipStream_t stream) {
  (void)in_sizes; (void)n_in; (void)out_size; (void)ws_size;

  int cur = 0;
  auto nx = [&]() -> const float* { return (const float*)d_in[cur++]; };

  // -------- unflatten parameters (setup_inputs dict order) --------
  const float *f1 = nx(), *f2 = nx(), *f3 = nx();
  const float *f1w = nx(), *f1b = nx(), *f2w = nx(), *f2b = nx(),
              *f3w = nx(), *f3b = nx();
  const float *a1w1=nx(), *a1b1=nx(), *a1w2=nx(), *a1b2=nx();
  const float *a2w1=nx(), *a2b1=nx(), *a2w2=nx(), *a2b2=nx();
  const float *aow=nx(), *aob=nx();
  static BLK blks[3][6];
  static DCP dcs[3];
  for (int g = 0; g < 3; ++g) {
    for (int i = 0; i < 6; ++i) {
      BLK& p = blks[g][i];
      p.ln_in_g = nx(); p.ln_in_b = nx();
      SSP& s = p.ss;
      s.in_w=nx(); s.conv_w=nx(); s.conv_b=nx(); s.xproj_w=nx();
      s.dt_w=nx(); s.dt_b=nx(); s.A_log=nx(); s.D=nx();
      s.on_g=nx(); s.on_b=nx(); s.out_w=nx();
      p.ln_ffn_g=nx(); p.ln_ffn_b=nx();
      p.fc1_w=nx(); p.fc1_b=nx(); p.fc2_w=nx(); p.fc2_b=nx();
      p.scale1=nx(); p.scale2=nx();
    }
    DCP& d = dcs[g];
    d.w1=nx(); d.b1=nx(); d.w2=nx(); d.b2=nx(); d.w3=nx(); d.b3=nx();
    d.w4=nx(); d.b4=nx(); d.w5=nx(); d.b5=nx();
  }
  const float *cab_w=nx(), *cab_b=nx();
  const float *ln1_g=nx(), *ln1_b=nx(), *ln2_g=nx(), *ln2_b=nx();
  const float *s1w1=nx(), *s1b1=nx(), *s1w2=nx(), *s1b2=nx();
  const float *sca_w=nx(), *sca_b=nx(), *mid_w=nx(), *mid_b=nx();
  const float *c1w=nx(), *c1b=nx(), *c2w=nx(), *c2b=nx();
  const float *ocw1=nx(), *ocb1=nx(), *ocw2=nx(), *ocb2=nx();
  const float *alw=nx(), *alb=nx();

  // -------- workspace partition (~60 MB of f32) --------
  float* ws = (float*)d_ws;
  const long T64  = (long)B_ * DM_ * L_;
  const long T128 = (long)B_ * DI_ * L_;
  float *F1C = ws,          *F2C = F1C + T64,  *F3C = F2C + T64;
  float *FUSED = F3C + T64, *Xb = FUSED + T64, *GI = Xb + T64;
  float *A64 = GI + T64,    *B64 = A64 + T64,  *C64 = B64 + T64;
  float *P128a = C64 + T64, *P128b = P128a + T128, *P128c = P128b + T128;
  float *XS  = P128c + T128;                    // (B,4,DI,L) — also ys in place
  float *DBL = XS + (long)B_ * 4 * DI_ * L_;    // (B,4,36,L)
  float *T192 = DBL + (long)B_ * 4 * 36 * L_;   // (B,192,L)
  float *SM0 = T192 + (long)B_ * 192 * L_;
  float *SM1 = SM0 + B_ * DM_;

  // -------- launch helpers --------
  auto gemmL = [&](int WT, const float* X, const float* Wp, int wld,
                   const float* bias, float* Y, int Cin, int Cout,
                   long xbs, long ybs, long wbs, int nw, int nb,
                   int act, float slope) {
    dim3 grid((unsigned)((Cout + 63) / 64), L_ / 32, (unsigned)nb);
    dim3 blk(32, 4, 2);
    const bool fast = (Cout % 16 == 0) && (Cin % 32 == 0);
    if (WT) {
      if (fast) mh_gemm_wmma<1, 0><<<grid, blk, 0, stream>>>(X, Wp, bias, Y, Cin, Cout,
                                                             wld, xbs, ybs, wbs, nw, act, slope);
      else      mh_gemm_wmma<1, 1><<<grid, blk, 0, stream>>>(X, Wp, bias, Y, Cin, Cout,
                                                             wld, xbs, ybs, wbs, nw, act, slope);
    } else {
      if (fast) mh_gemm_wmma<0, 0><<<grid, blk, 0, stream>>>(X, Wp, bias, Y, Cin, Cout,
                                                             wld, xbs, ybs, wbs, nw, act, slope);
      else      mh_gemm_wmma<0, 1><<<grid, blk, 0, stream>>>(X, Wp, bias, Y, Cin, Cout,
                                                             wld, xbs, ybs, wbs, nw, act, slope);
    }
  };
  auto lin = [&](int WT, const float* X, const float* Wp, int wld,
                 const float* bias, float* Y, int Cin, int Cout,
                 int act, float slope) {
    gemmL(WT, X, Wp, wld, bias, Y, Cin, Cout,
          (long)Cin * L_, (long)Cout * L_, 0, 1, B_, act, slope);
  };
  auto conv = [&](const float* X, const float* Wp, const float* bias, float* Y,
                  int Cin, int Cout, int K, int pad, int dil, int groups,
                  int act, float slope) {
    long total = (long)B_ * Cout * L_;
    mh_conv2d<<<dim3((unsigned)((total + 255) / 256)), 256, 0, stream>>>(
        X, Wp, bias, Y, Cin, Cout, K, pad, dil, groups, act, slope);
  };
  auto lnc = [&](const float* X, const float* g, const float* b, float* Y,
                 int C, float eps) {
    long total = (long)B_ * L_;
    mh_ln_ch<<<dim3((unsigned)((total + 255) / 256)), 256, 0, stream>>>(X, g, b, Y, C, eps);
  };
  auto gblk = [](long n) { return dim3((unsigned)((n + 255) / 256)); };

  // -------- input stem + alignment --------
  conv(f1, f1w, f1b, F1C, 6, DM_, 3, 1, 1, 1, 0, 0.f);
  conv(f2, f2w, f2b, F2C, 6, DM_, 3, 1, 1, 1, 0, 0.f);
  conv(f3, f3w, f3b, F3C, 6, DM_, 3, 1, 1, 1, 0, 0.f);

  long n128 = (long)B_ * 128 * L_, n192 = (long)B_ * 192 * L_;
  long n64 = T64;
  mh_concat2<<<gblk(n128), 256, 0, stream>>>(F1C, F2C, P128a);
  conv(P128a, a1w1, a1b1, P128b, 128, 128, 3, 1, 1, 1, 1, 0.01f);
  conv(P128b, a1w2, a1b2, A64, 128, 64, 3, 1, 1, 1, 3, 0.f);      // a12
  mh_concat2<<<gblk(n128), 256, 0, stream>>>(F3C, F2C, P128a);
  conv(P128a, a2w1, a2b1, P128b, 128, 128, 3, 1, 1, 1, 1, 0.01f);
  conv(P128b, a2w2, a2b2, B64, 128, 64, 3, 1, 1, 1, 3, 0.f);      // a32
  mh_cat3<<<gblk(n192), 256, 0, stream>>>(F1C, A64, F2C, F3C, B64, T192);
  lin(1, T192, aow, 192, aob, FUSED, 192, DM_, 0, 0.f);           // 1x1 fuse
  mh_copy<<<gblk(n64), 256, 0, stream>>>(FUSED, Xb, n64);

  // -------- one basical block --------
  auto run_block = [&](const BLK& p) {
    lnc(Xb, p.ln_in_g, p.ln_in_b, A64, DM_, 1e-6f);               // ln2d
    // ss4d on A64 -> B64
    lin(0, A64, p.ss.in_w,        2 * DI_, nullptr, P128a, DM_, DI_, 0, 0.f); // xi
    lin(0, A64, p.ss.in_w + DI_,  2 * DI_, nullptr, P128b, DM_, DI_, 0, 0.f); // z
    conv(P128a, p.ss.conv_w, p.ss.conv_b, P128c, DI_, DI_, 3, 1, 1, DI_, 2, 0.f); // dw+silu
    mh_build_xs<<<gblk((long)B_ * 4 * DI_ * L_), 256, 0, stream>>>(P128c, XS);
    gemmL(1, XS, p.ss.xproj_w, DI_, nullptr, DBL, DI_, 36,
          (long)DI_ * L_, 36L * L_, 36L * DI_, 4, B_ * 4, 0, 0.f);
    mh_scan<<<dim3(B_ * 4), 128, 0, stream>>>(XS, DBL, p.ss.dt_w, p.ss.dt_b,
                                              p.ss.A_log, p.ss.D);
    mh_combine<<<gblk((long)B_ * DI_ * L_), 256, 0, stream>>>(XS, P128a);
    lnc(P128a, p.ss.on_g, p.ss.on_b, P128a, DI_, 1e-5f);
    mh_mul_silu<<<gblk(n128), 256, 0, stream>>>(P128a, P128b, P128a, n128);
    lin(0, P128a, p.ss.out_w, DM_, nullptr, B64, DI_, DM_, 0, 0.f);
    // x = ss(ln(x)) + x*scale1
    mh_addscale<<<gblk(n64), 256, 0, stream>>>(B64, Xb, p.scale1, Xb, DM_, n64);
    // FFN
    lnc(Xb, p.ln_ffn_g, p.ln_ffn_b, A64, DM_, 1e-5f);
    lin(0, A64, p.fc1_w, 2 * DM_, p.fc1_b, P128a, DM_, 2 * DM_, 1, 0.01f);
    lin(0, P128a, p.fc2_w, DM_, p.fc2_b, B64, 2 * DM_, DM_, 0, 0.f);
    mh_addscale<<<gblk(n64), 256, 0, stream>>>(B64, Xb, p.scale2, Xb, DM_, n64);
  };

  // -------- dilated conv group tail --------
  auto run_dilated = [&](const DCP& d) {
    gemmL(1, Xb, d.w1, DM_, d.b1, A64, DM_, 16, (long)DM_ * L_, 16L * L_, 0, 1, B_, 0, 0.f);
    conv(A64, d.w2, d.b2, B64, 16, 16, 3, 2, 2, 16, 1, 0.2f);
    gemmL(1, B64, d.w3, 16, d.b3, A64, 16, 16, 16L * L_, 16L * L_, 0, 1, B_, 1, 0.2f);
    gemmL(1, A64, d.w4, 16, d.b4, B64, 16, DM_, 16L * L_, (long)DM_ * L_, 0, 1, B_, 0, 0.f);
    conv(B64, d.w5, d.b5, A64, DM_, DM_, 3, 2, 2, DM_, 0, 0.f);
    mh_add<<<gblk(n64), 256, 0, stream>>>(A64, GI, Xb, n64);
  };

  for (int g = 0; g < 3; ++g) {
    mh_copy<<<gblk(n64), 256, 0, stream>>>(Xb, GI, n64);
    for (int i = 0; i < 6; ++i) run_block(blks[g][i]);
    run_dilated(dcs[g]);
  }

  // -------- tail: cab conv + postproc + output convs --------
  mh_add<<<gblk(n64), 256, 0, stream>>>(Xb, FUSED, A64, n64);
  lin(1, A64, cab_w, DM_, cab_b, Xb, DM_, DM_, 0, 0.f);

  // postproc (NAFBlock-style)
  lnc(Xb, ln1_g, ln1_b, A64, DM_, 1e-6f);
  lin(1, A64, s1w1, DM_, s1b1, P128a, DM_, 128, 0, 0.f);
  conv(P128a, s1w2, s1b2, P128c, 128, 128, 3, 1, 1, DM_, 0, 0.f);
  mh_halfmul<<<gblk(n64), 256, 0, stream>>>(P128c, A64);          // x1*x2
  mh_smean<<<dim3(B_ * DM_), 256, 0, stream>>>(A64, SM0);
  mh_slinear<<<1, 128, 0, stream>>>(SM0, sca_w, sca_b, SM1);
  mh_mul_bc<<<gblk(n64), 256, 0, stream>>>(A64, SM1, DM_);
  lin(1, A64, mid_w, DM_, mid_b, B64, DM_, DM_, 0, 0.f);
  mh_add<<<gblk(n64), 256, 0, stream>>>(B64, Xb, C64, n64);       // y
  lnc(C64, ln2_g, ln2_b, A64, DM_, 1e-6f);
  lin(1, A64, c1w, DM_, c1b, P128a, DM_, 128, 0, 0.f);
  mh_halfmul<<<gblk(n64), 256, 0, stream>>>(P128a, A64);
  lin(1, A64, c2w, DM_, c2b, B64, DM_, DM_, 0, 0.f);
  mh_add<<<gblk(n64), 256, 0, stream>>>(B64, C64, Xb, n64);

  // output convs + final sigmoid
  lin(1, Xb, ocw1, DM_, ocb1, A64, DM_, DM_, 0, 0.f);
  conv(A64, ocw2, ocb2, B64, DM_, DM_, 3, 1, 1, DM_, 0, 0.f);
  mh_add<<<gblk(n64), 256, 0, stream>>>(B64, F2C, A64, n64);
  conv(A64, alw, alb, (float*)d_out, DM_, 3, 3, 1, 1, 1, 3, 0.f);
}
